// SelfAttention_37391985279526
// MI455X (gfx1250) — matrix operand order repaired and checked by reading, not compiled, not run on hardware
//
#include <hip/hip_runtime.h>
#include <hip/hip_bf16.h>

// ---------------------------------------------------------------------------
// Self-attention (B=2, N=2048, D=1024, H=16, hd=64), fp32 in/out.
// All GEMMs run on v_wmma_f32_16x16x32_bf16 (bf16 inputs, f32 accumulate).
// ---------------------------------------------------------------------------

typedef __bf16 bf16_t;
typedef __attribute__((ext_vector_type(16))) __bf16 v16bf;
typedef __attribute__((ext_vector_type(8)))  __bf16 v8bf;
typedef __attribute__((ext_vector_type(8)))  float  v8f;

#define DIMX   1024
#define NHEAD  16
#define HD     64
#define BATCH  2
#define SEQ    2048
#define TOKENS (BATCH * SEQ)   // 4096
#define QKVN   (3 * DIMX)      // 3072

__device__ __forceinline__ bf16_t to_bf16(float f) { return (bf16_t)f; }

// A-matrix tile (16x32 bf16, MxK) from row-major src (lda in elements).
// Contiguous-k layout used here: lane = m + 16*kh; VGPR0-3 = K kh*16+0..7, VGPR4-7 = K kh*16+8..15.
__device__ __forceinline__ v16bf load_a_tile(const bf16_t* base, int lda, int lane) {
    int m  = lane & 15;
    int kh = lane >> 4;
    const v8bf lo = *(const v8bf*)(base + (size_t)m * lda + kh * 16);
    const v8bf hi = *(const v8bf*)(base + (size_t)m * lda + kh * 16 + 8);
    v16bf a;
#pragma unroll
    for (int i = 0; i < 8; ++i) { a[i] = lo[i]; a[8 + i] = hi[i]; }
    return a;
}

// B-matrix tile (32x16 bf16, KxN) from a TRANSPOSED row-major source
// baseT[n][k] (ldb in elements). Contiguous-k layout used here: lane = n + 16*kh; VGPR i holds
// K = kh*16 + 2i, 2i+1 for column n -> 16 consecutive K = one 32B load.
__device__ __forceinline__ v16bf load_b_tile(const bf16_t* baseT, int ldb, int lane) {
    int n  = lane & 15;
    int kh = lane >> 4;
    return *(const v16bf*)(baseT + (size_t)n * ldb + kh * 16);
}

__device__ __forceinline__ v8f wmma_bf16(v16bf a, v16bf b, v8f c) {
    return __builtin_amdgcn_wmma_f32_16x16x32_bf16(
        /*neg_a=*/false, a, /*neg_b=*/false, b,
        /*c_mod=*/(short)0, c, /*reuse_a=*/false, /*reuse_b=*/false);
}

// ---------------------------------------------------------------------------
// Kernel 1a: fp32 -> bf16 straight convert
// ---------------------------------------------------------------------------
__global__ void k_cvt(const float* __restrict__ in, bf16_t* __restrict__ out, int n) {
    int i = blockIdx.x * blockDim.x + threadIdx.x;
    if (i < n) out[i] = to_bf16(in[i]);
}

// Kernel 1b: fp32 [K x N] row-major -> bf16 transposed [N x K] row-major
__global__ void k_cvt_t(const float* __restrict__ in, bf16_t* __restrict__ outT,
                        int K, int N) {
    int i = blockIdx.x * blockDim.x + threadIdx.x;
    if (i < K * N) {
        int n = i / K;
        int k = i - n * K;
        outT[i] = to_bf16(in[(size_t)k * N + n]);
    }
}

// ---------------------------------------------------------------------------
// Kernel 2: QKV = Xb(4096x1024) @ WqkvT + bias, scattered into per-head
//   Q[b,h,n,d], K[b,h,n,d] (row-major over d) and V^T[b,h,d,n].
// Each wave owns a 64(M) x 64(N) block: 4x4 accumulator tiles
// -> 16 WMMAs per 16 b128 loads per K-step.
// ---------------------------------------------------------------------------
__global__ __launch_bounds__(256) void k_qkv(const bf16_t* __restrict__ xb,
                                             const bf16_t* __restrict__ wT,
                                             const float*  __restrict__ bias,
                                             bf16_t* __restrict__ qh,
                                             bf16_t* __restrict__ kh_,
                                             bf16_t* __restrict__ vT) {
    const int lane = threadIdx.x & 31;
    const int wave = blockIdx.x * (blockDim.x >> 5) + (threadIdx.x >> 5);
    const int numNB = QKVN / 64;              // 48
    const int mb = wave / numNB;              // 0..63
    const int nb = wave - mb * numNB;         // 0..47
    const int m0 = mb * 64;
    const int n0 = nb * 64;

    v8f acc[4][4] = {};

    for (int kb = 0; kb < DIMX; kb += 32) {
        v16bf a[4];
#pragma unroll
        for (int mt = 0; mt < 4; ++mt)
            a[mt] = load_a_tile(xb + (size_t)(m0 + mt * 16) * DIMX + kb, DIMX, lane);
#pragma unroll
        for (int nt = 0; nt < 4; ++nt) {
            v16bf b = load_b_tile(wT + (size_t)(n0 + nt * 16) * DIMX + kb, DIMX, lane);
#pragma unroll
            for (int mt = 0; mt < 4; ++mt)
                acc[mt][nt] = wmma_bf16(a[mt], b, acc[mt][nt]);
        }
    }

    const int nlane = lane & 15;
    const int hi    = lane >> 4;
#pragma unroll
    for (int mt = 0; mt < 4; ++mt) {
#pragma unroll
        for (int nt = 0; nt < 4; ++nt) {
            int f   = n0 + nt * 16 + nlane;       // 0..3071
            float bv = bias[f];
            int sel = f >> 10;                    // 0=q 1=k 2=v
            int rem = f & 1023;
            int h   = rem >> 6;
            int d   = rem & 63;
#pragma unroll
            for (int r = 0; r < 8; ++r) {
                int m    = m0 + mt * 16 + r + 8 * hi;  // global token row
                int bi   = m >> 11;
                int tok  = m & 2047;
                size_t hb = (size_t)(bi * NHEAD + h);
                bf16_t val = to_bf16(acc[mt][nt][r] + bv);
                if (sel == 0)      qh [(hb * SEQ + tok) * HD + d] = val;
                else if (sel == 1) kh_[(hb * SEQ + tok) * HD + d] = val;
                else               vT [(hb * HD + d) * SEQ + tok] = val;
            }
        }
    }
}

// ---------------------------------------------------------------------------
// Kernel 3: flash attention. One wave handles 32 queries (2 row tiles) of one
// (b,h), streaming keys in chunks of 32 with online softmax. K/V tiles are
// loaded once per chunk and reused across both query tiles. P is re-laid-out
// from C-layout to A-layout through per-wave LDS buffers.
// ---------------------------------------------------------------------------
__global__ __launch_bounds__(128) void k_attn(const bf16_t* __restrict__ qh,
                                              const bf16_t* __restrict__ kh_,
                                              const bf16_t* __restrict__ vT,
                                              bf16_t* __restrict__ ob) {
    __shared__ bf16_t pshare[4][2][16 * 32];

    const int lane = threadIdx.x & 31;
    const int wid  = threadIdx.x >> 5;
    const int task = blockIdx.x * 4 + wid;     // 0..2047
    const int qblk = task & 63;                // 32-query block within (b,h)
    const int bh   = task >> 6;                // 0..31

    const bf16_t* qbase = qh  + ((size_t)bh * SEQ + (size_t)qblk * 32) * HD;
    const bf16_t* kbase = kh_ + (size_t)bh * SEQ * HD;
    const bf16_t* vbase = vT  + (size_t)bh * HD * SEQ;

    // Q A-tiles kept resident: [query tile][d half]
    v16bf qa[2][2];
#pragma unroll
    for (int qt = 0; qt < 2; ++qt) {
        qa[qt][0] = load_a_tile(qbase + (size_t)qt * 16 * HD,      HD, lane);
        qa[qt][1] = load_a_tile(qbase + (size_t)qt * 16 * HD + 32, HD, lane);
    }

    v8f o[2][4] = {};
    float mrow[2][8], lrow[2][8];
#pragma unroll
    for (int qt = 0; qt < 2; ++qt)
#pragma unroll
        for (int r = 0; r < 8; ++r) { mrow[qt][r] = -1.0e30f; lrow[qt][r] = 0.0f; }

    const int nlane = lane & 15;
    const int hi    = lane >> 4;
    const float scale = 0.125f;                // 1/sqrt(64)

    for (int kc = 0; kc < SEQ; kc += 32) {
        // ---- K tiles for this chunk (shared by both query tiles)
        v16bf b0 = load_b_tile(kbase + (size_t)kc * HD,             HD, lane);
        v16bf b1 = load_b_tile(kbase + (size_t)kc * HD + 32,        HD, lane);
        v16bf b2 = load_b_tile(kbase + (size_t)(kc + 16) * HD,      HD, lane);
        v16bf b3 = load_b_tile(kbase + (size_t)(kc + 16) * HD + 32, HD, lane);

        // ---- scores: per query tile, two 16q x 16k tiles (K-dim = 64)
        v8f s[2][2] = {};
#pragma unroll
        for (int qt = 0; qt < 2; ++qt) {
            s[qt][0] = wmma_bf16(qa[qt][0], b0, s[qt][0]);
            s[qt][0] = wmma_bf16(qa[qt][1], b1, s[qt][0]);
            s[qt][1] = wmma_bf16(qa[qt][0], b2, s[qt][1]);
            s[qt][1] = wmma_bf16(qa[qt][1], b3, s[qt][1]);
        }

        // ---- online softmax per row (VGPR r -> row r + 8*hi, lane&15 -> col)
#pragma unroll
        for (int qt = 0; qt < 2; ++qt) {
            bf16_t* lds = pshare[wid][qt];
#pragma unroll
            for (int r = 0; r < 8; ++r) {
                float x0 = s[qt][0][r] * scale;
                float x1 = s[qt][1][r] * scale;
                float mx = fmaxf(x0, x1);
                mx = fmaxf(mx, __shfl_xor(mx, 1, 32));
                mx = fmaxf(mx, __shfl_xor(mx, 2, 32));
                mx = fmaxf(mx, __shfl_xor(mx, 4, 32));
                mx = fmaxf(mx, __shfl_xor(mx, 8, 32));
                float mnew  = fmaxf(mrow[qt][r], mx);
                float alpha = __expf(mrow[qt][r] - mnew);
                mrow[qt][r] = mnew;
                float p0 = __expf(x0 - mnew);
                float p1 = __expf(x1 - mnew);
                float rs = p0 + p1;
                rs += __shfl_xor(rs, 1, 32);
                rs += __shfl_xor(rs, 2, 32);
                rs += __shfl_xor(rs, 4, 32);
                rs += __shfl_xor(rs, 8, 32);
                lrow[qt][r] = lrow[qt][r] * alpha + rs;
                o[qt][0][r] *= alpha; o[qt][1][r] *= alpha;
                o[qt][2][r] *= alpha; o[qt][3][r] *= alpha;
                // stage P (C layout -> LDS row-major 16x32)
                lds[(r + 8 * hi) * 32 + nlane]      = to_bf16(p0);
                lds[(r + 8 * hi) * 32 + 16 + nlane] = to_bf16(p1);
            }
        }

        asm volatile("s_wait_dscnt 0" ::: "memory");  // P visible across lanes

        // ---- O += P(16x32) @ V(32x64): V tiles shared by both query tiles
        v16bf pa0 = load_a_tile(pshare[wid][0], 32, lane);
        v16bf pa1 = load_a_tile(pshare[wid][1], 32, lane);
#pragma unroll
        for (int db = 0; db < 4; ++db) {
            v16bf vb = load_b_tile(vbase + (size_t)(db * 16) * SEQ + kc, SEQ, lane);
            o[0][db] = wmma_bf16(pa0, vb, o[0][db]);
            o[1][db] = wmma_bf16(pa1, vb, o[1][db]);
        }
    }

    // ---- normalize and write attn output as bf16 [token][feature]
    const int h  = bh & 15;
    const int bi = bh >> 4;
#pragma unroll
    for (int qt = 0; qt < 2; ++qt) {
        bf16_t* obase = ob + ((size_t)bi * SEQ + (size_t)qblk * 32 + qt * 16) * DIMX
                        + h * HD;
#pragma unroll
        for (int r = 0; r < 8; ++r) {
            float inv = 1.0f / lrow[qt][r];
            int   row = r + 8 * hi;
#pragma unroll
            for (int db = 0; db < 4; ++db) {
                int d = db * 16 + nlane;
                obase[(size_t)row * DIMX + d] = to_bf16(o[qt][db][r] * inv);
            }
        }
    }
}

// ---------------------------------------------------------------------------
// Kernel 4: out = Attn(4096x1024) @ WoutT + bias  -> fp32
// 64x64 block per wave, 4x4 accumulator tiles.
// ---------------------------------------------------------------------------
__global__ __launch_bounds__(256) void k_out(const bf16_t* __restrict__ ab,
                                             const bf16_t* __restrict__ wT,
                                             const float*  __restrict__ bias,
                                             float* __restrict__ out) {
    const int lane = threadIdx.x & 31;
    const int wave = blockIdx.x * (blockDim.x >> 5) + (threadIdx.x >> 5);
    const int numNB = DIMX / 64;              // 16
    const int mb = wave / numNB;              // 0..63
    const int nb = wave - mb * numNB;
    const int m0 = mb * 64;
    const int n0 = nb * 64;

    v8f acc[4][4] = {};

    for (int kb = 0; kb < DIMX; kb += 32) {
        v16bf a[4];
#pragma unroll
        for (int mt = 0; mt < 4; ++mt)
            a[mt] = load_a_tile(ab + (size_t)(m0 + mt * 16) * DIMX + kb, DIMX, lane);
#pragma unroll
        for (int nt = 0; nt < 4; ++nt) {
            v16bf b = load_b_tile(wT + (size_t)(n0 + nt * 16) * DIMX + kb, DIMX, lane);
#pragma unroll
            for (int mt = 0; mt < 4; ++mt)
                acc[mt][nt] = wmma_bf16(a[mt], b, acc[mt][nt]);
        }
    }

    const int nlane = lane & 15;
    const int hi    = lane >> 4;
#pragma unroll
    for (int mt = 0; mt < 4; ++mt) {
#pragma unroll
        for (int nt = 0; nt < 4; ++nt) {
            int n = n0 + nt * 16 + nlane;
            float bv = bias[n];
#pragma unroll
            for (int r = 0; r < 8; ++r) {
                int m = m0 + mt * 16 + r + 8 * hi;
                out[(size_t)m * DIMX + n] = acc[mt][nt][r] + bv;
            }
        }
    }
}

// ---------------------------------------------------------------------------
extern "C" void kernel_launch(void* const* d_in, const int* in_sizes, int n_in,
                              void* d_out, int out_size, void* d_ws, size_t ws_size,
                              hipStream_t stream) {
    const float* x      = (const float*)d_in[0];
    const float* w_qkv  = (const float*)d_in[1];
    const float* b_qkv  = (const float*)d_in[2];
    const float* w_out  = (const float*)d_in[3];
    const float* b_out  = (const float*)d_in[4];
    float*       out    = (float*)d_out;

    char*  ws  = (char*)d_ws;
    size_t off = 0;
    auto alloc = [&](size_t bytes) -> void* {
        off = (off + 255) & ~(size_t)255;
        void* p = ws + off;
        off += bytes;
        return p;
    };

    bf16_t* xb     = (bf16_t*)alloc((size_t)TOKENS * DIMX * 2);   // 8 MB
    bf16_t* wqkvT  = (bf16_t*)alloc((size_t)QKVN * DIMX * 2);     // 6 MB
    bf16_t* woT    = (bf16_t*)alloc((size_t)DIMX * DIMX * 2);     // 2 MB
    bf16_t* qbuf   = (bf16_t*)alloc((size_t)TOKENS * DIMX * 2);   // 8 MB
    bf16_t* kbuf   = (bf16_t*)alloc((size_t)TOKENS * DIMX * 2);   // 8 MB
    bf16_t* vTbuf  = (bf16_t*)alloc((size_t)TOKENS * DIMX * 2);   // 8 MB
    bf16_t* attnb  = (bf16_t*)alloc((size_t)TOKENS * DIMX * 2);   // 8 MB
    (void)ws_size; (void)in_sizes; (void)n_in; (void)out_size;

    // 1) precision conversion / weight transposes
    k_cvt  <<<(TOKENS * DIMX + 255) / 256, 256, 0, stream>>>(x, xb, TOKENS * DIMX);
    k_cvt_t<<<(DIMX * QKVN + 255) / 256, 256, 0, stream>>>(w_qkv, wqkvT, DIMX, QKVN);
    k_cvt_t<<<(DIMX * DIMX + 255) / 256, 256, 0, stream>>>(w_out, woT, DIMX, DIMX);

    // 2) QKV projection: 64 * 48 = 3072 wave-blocks, 8 waves per block
    k_qkv<<<(64 * 48) / 8, 256, 0, stream>>>(xb, wqkvT, b_qkv, qbuf, kbuf, vTbuf);

    // 3) attention: 32 (b,h) * 64 q-blocks = 2048 wave tasks, 4 waves/block
    k_attn<<<2048 / 4, 128, 0, stream>>>(qbuf, kbuf, vTbuf, attnb);

    // 4) output projection: 64 * 16 = 1024 wave-blocks
    k_out<<<(64 * 16) / 8, 256, 0, stream>>>(attnb, woT, b_out, out);
}
